// MogLSTM_28827820491423
// MI455X (gfx1250) — matrix operand order repaired and checked by reading, not compiled, not run on hardware
//
#include <hip/hip_runtime.h>
#include <hip/hip_bf16.h>

// ---------------------------------------------------------------------------
// Mogrifier bi-LSTM, B=1, T=4096, I=H=1024 on gfx1250 (MI455X).
//
// _mogrify collapses: ct is loop-invariant, so xt <- (2*tanh(c@Q+b))^5 * xt.
// Per step/direction: GEMV c@Qt (1024x1024) then GEMV [xg,h]@[Wih;Whh]^T
// (2048x4096). Weights converted once to bf16 (~34MB, L2-resident). Each
// direction split over 16 WGs (L2 streaming bw), 2 global barriers/step.
// GEMV via v_wmma_f32_16x16x32_bf16: A = 16 weight rows x 32 K (non-temporal
// global b128 loads), B = activation vector broadcast into 16 cols (LDS).
// f32 cell state lives in registers of its owning thread for all 4096 steps;
// transcendentals are branchless v_exp/v_rcp forms.
// ---------------------------------------------------------------------------

typedef __attribute__((ext_vector_type(16))) __bf16        v16bf;
typedef __attribute__((ext_vector_type(8)))  float         v8f;
typedef __attribute__((ext_vector_type(4)))  unsigned int  u32x4;

union BF16Tile {
  v16bf v;
  u32x4 q[2];
};

#define HID   1024
#define TLEN  4096
#define NWG   16      // workgroups per direction
#define NTHR  256     // threads per workgroup (8 wave32s)

__device__ __forceinline__ unsigned short f2bf(float f) {
  unsigned int u = __float_as_uint(f);
  unsigned int r = u + 0x7FFFu + ((u >> 16) & 1u);  // round-to-nearest-even
  return (unsigned short)(r >> 16);
}

// Branchless hardware transcendentals: v_exp_f32 + v_rcp_f32.
__device__ __forceinline__ float sig_fast(float x) {
  return __builtin_amdgcn_rcpf(1.0f + __expf(-x));
}
__device__ __forceinline__ float tanh_fast(float x) {
  // tanh(x) = 1 - 2/(exp(2x)+1); saturates correctly as exp -> 0 or inf.
  return 1.0f - 2.0f * __builtin_amdgcn_rcpf(__expf(2.0f * x) + 1.0f);
}

// Single-tile GEMV MAC: acc += W[rb:rb+16, 0:nk] * vec (vec broadcast in B).
// A per-lane (16x32 bf16): lo lanes K{0..7,16..23}, hi lanes K{8..15,24..31}.
// B per-lane: same K order as A, lo lanes K{0..7,16..23}, hi lanes K{8..15,24..31}.
__device__ __forceinline__ v8f bf16_gemv_tile1(const unsigned short* __restrict__ W,
                                               int rb,
                                               const unsigned short* __restrict__ vecp,
                                               int nk, int lane, v8f acc) {
  const int m  = lane & 15;
  const int oA = (lane >> 4) * 8;
  const int oB = (lane >> 4) * 8;
  const unsigned short* wr = W + (size_t)(rb + m) * HID + oA;
  const unsigned short* vp = vecp + oB;
#pragma unroll 4
  for (int k0 = 0; k0 < nk; k0 += 32) {
    BF16Tile a, b;
    b.q[0] = *(const u32x4*)(vp + k0);
    b.q[1] = *(const u32x4*)(vp + k0 + 16);
    a.q[0] = __builtin_nontemporal_load((const u32x4*)(wr + k0));
    a.q[1] = __builtin_nontemporal_load((const u32x4*)(wr + k0 + 16));
    acc = __builtin_amdgcn_wmma_f32_16x16x32_bf16(false, a.v, false, b.v,
                                                  (short)0, acc, false, false);
  }
  return acc;
}

// Dual-tile GEMV MAC sharing the B operand: two output tiles, one vec load.
__device__ __forceinline__ void bf16_gemv_tile2(const unsigned short* __restrict__ W,
                                                int rb0, int rb1,
                                                const unsigned short* __restrict__ vecp,
                                                int nk, int lane,
                                                v8f& acc0, v8f& acc1) {
  const int m  = lane & 15;
  const int oA = (lane >> 4) * 8;
  const int oB = (lane >> 4) * 8;
  const unsigned short* wr0 = W + (size_t)(rb0 + m) * HID + oA;
  const unsigned short* wr1 = W + (size_t)(rb1 + m) * HID + oA;
  const unsigned short* vp  = vecp + oB;
#pragma unroll 2
  for (int k0 = 0; k0 < nk; k0 += 32) {
    BF16Tile a0, a1, b;
    b.q[0] = *(const u32x4*)(vp + k0);
    b.q[1] = *(const u32x4*)(vp + k0 + 16);
    a0.q[0] = __builtin_nontemporal_load((const u32x4*)(wr0 + k0));
    a0.q[1] = __builtin_nontemporal_load((const u32x4*)(wr0 + k0 + 16));
    a1.q[0] = __builtin_nontemporal_load((const u32x4*)(wr1 + k0));
    a1.q[1] = __builtin_nontemporal_load((const u32x4*)(wr1 + k0 + 16));
    acc0 = __builtin_amdgcn_wmma_f32_16x16x32_bf16(false, a0.v, false, b.v,
                                                   (short)0, acc0, false, false);
    acc1 = __builtin_amdgcn_wmma_f32_16x16x32_bf16(false, a1.v, false, b.v,
                                                   (short)0, acc1, false, false);
  }
}

// Monotonic arrive-counter barrier across the 16 WGs of one direction.
__device__ __forceinline__ void grid_barrier(unsigned int* cnt, unsigned int target) {
  __threadfence();
  __syncthreads();
  if (threadIdx.x == 0) {
    __hip_atomic_fetch_add(cnt, 1u, __ATOMIC_RELEASE, __HIP_MEMORY_SCOPE_AGENT);
    while (__hip_atomic_load(cnt, __ATOMIC_ACQUIRE, __HIP_MEMORY_SCOPE_AGENT) < target) {
      __builtin_amdgcn_s_sleep(2);
    }
  }
  __syncthreads();
  __threadfence();
}

// ---------------------- weight prep (run once per launch) ------------------

__global__ void k_prep_w(const float* __restrict__ Wih_f, const float* __restrict__ Whh_f,
                         const float* __restrict__ Wih_b, const float* __restrict__ Whh_b,
                         unsigned short* __restrict__ WihBf2,
                         unsigned short* __restrict__ WhhBf2) {
  const size_t N = (size_t)4 * HID * HID;
  size_t i = (size_t)blockIdx.x * blockDim.x + threadIdx.x;
  if (i < N) {
    WihBf2[i]     = f2bf(Wih_f[i]);
    WihBf2[N + i] = f2bf(Wih_b[i]);
    WhhBf2[i]     = f2bf(Whh_f[i]);
    WhhBf2[N + i] = f2bf(Whh_b[i]);
  }
}

__global__ void k_prep_misc(const float* __restrict__ Q,
                            const float* __restrict__ bihf, const float* __restrict__ bhhf,
                            const float* __restrict__ bihb, const float* __restrict__ bhhb,
                            unsigned short* __restrict__ QtBf,
                            float* __restrict__ bias2,
                            unsigned short* __restrict__ cbf2,
                            unsigned short* __restrict__ hbf2,
                            unsigned short* __restrict__ xg2,
                            unsigned int* __restrict__ cnts) {
  size_t i = (size_t)blockIdx.x * blockDim.x + threadIdx.x;
  if (i < (size_t)HID * HID) {   // Qt = Q^T so GEMV reads contiguous rows
    int r = (int)(i >> 10), c = (int)(i & (HID - 1));
    QtBf[(size_t)c * HID + r] = f2bf(Q[i]);
  }
  if (i < 4 * HID) {
    bias2[i]           = bihf[i] + bhhf[i];
    bias2[4 * HID + i] = bihb[i] + bhhb[i];
    hbf2[i] = 0;                 // 2 dirs x 2 buffers x 1024
  }
  if (i < 2 * HID) { cbf2[i] = 0; xg2[i] = 0; }
  if (i < 32) cnts[i] = 0;
}

// ---------------------- persistent recurrence kernel ------------------------

__global__ void __launch_bounds__(NTHR)
k_moglstm(const float* __restrict__ x,
          const unsigned short* __restrict__ WihBf2,
          const unsigned short* __restrict__ WhhBf2,
          const unsigned short* __restrict__ QtBf,
          const float* __restrict__ bias2,
          const float* __restrict__ cbias,
          unsigned short* __restrict__ cbf2,
          unsigned short* __restrict__ hbf2,
          unsigned short* __restrict__ xg2,
          unsigned int* __restrict__ cnts,
          float* __restrict__ out)
{
  const int dir  = blockIdx.x >> 4;
  const int wg   = blockIdx.x & 15;
  const int tid  = threadIdx.x;
  const int wave = tid >> 5;
  const int lane = tid & 31;

  const unsigned short* Wih = WihBf2 + (size_t)dir * 4 * HID * HID;
  const unsigned short* Whh = WhhBf2 + (size_t)dir * 4 * HID * HID;
  const float* bias = bias2 + dir * 4 * HID;
  unsigned short* cbf  = cbf2 + dir * HID;
  unsigned short* hbfD = hbf2 + dir * 2 * HID;
  unsigned short* xg   = xg2  + dir * HID;
  unsigned int* cnt    = cnts + dir * 16;

  __shared__ __attribute__((aligned(16))) unsigned short vec[2 * HID];
  __shared__ float gpre[4 * 64];   // phase1: [2][64] partials; phase2: [4][64]

  // f32 cell state lives in a register of its owning thread (tid<64) for the
  // whole 4096-step recurrence; only the bf16 copy goes through memory.
  float creg = 0.0f;
  unsigned int bar = 0;

  // Preload per-thread bias/cbias (loop-invariant) for the elementwise stage.
  float bi = 0.f, bf_ = 0.f, bg = 0.f, bo = 0.f, cb = 0.f;
  if (tid < 64) {
    const int j = wg * 64 + tid;
    bi  = bias[j];
    bf_ = bias[HID + j];
    bg  = bias[2 * HID + j];
    bo  = bias[3 * HID + j];
    cb  = cbias[j];
  }

  for (int t = 0; t < TLEN; ++t) {
    const int xrow = dir ? (TLEN - 1 - t) : t;
    const float* xt = x + (size_t)xrow * HID;
    const unsigned short* hbfR = hbfD + ((t)     & 1) * HID;  // read buffer
    unsigned short*       hbfW = hbfD + ((t + 1) & 1) * HID;  // write buffer

    // ---- Phase 1: mogrifier gate on x_t (skipped for t==0 per reference) --
    if (t == 0) {
      if (tid < 64) { int j = wg * 64 + tid; xg[j] = f2bf(xt[j]); }
    } else {
      unsigned int* v32 = (unsigned int*)vec;
      const unsigned int* c32 = (const unsigned int*)cbf;
      for (int i = tid; i < HID / 2; i += NTHR) v32[i] = c32[i];
      __syncthreads();
      // 8 waves = 4 tiles x 2 K-halves (K-split halves phase-1 latency)
      {
        const int tile = wave >> 1;
        const int kh   = wave & 1;
        const int rb   = wg * 64 + tile * 16;
        v8f acc = {};
        acc = bf16_gemv_tile1(QtBf + kh * (HID / 2), rb, vec + kh * (HID / 2),
                              HID / 2, lane, acc);
        if ((lane & 15) == 0) {
          const int hi8 = (lane >> 4) * 8;
#pragma unroll
          for (int r = 0; r < 8; ++r)
            gpre[kh * 64 + tile * 16 + hi8 + r] = acc[r];
        }
      }
      __syncthreads();
      if (tid < 64) {                          // parallel transcendental stage
        const int j = wg * 64 + tid;
        const float v = gpre[tid] + gpre[64 + tid] + cb;
        const float mfac = 2.0f * tanh_fast(v);
        float p = mfac * mfac;                 // ^2
        p = p * p * mfac;                      // ^5
        xg[j] = f2bf(p * xt[j]);
      }
    }
    grid_barrier(cnt, 16u * (++bar));          // all xg visible, vec reusable

    // ---- Phase 2: gates = [xg, h] @ [Wih ; Whh]^T, then cell update -------
    {
      unsigned int* v32 = (unsigned int*)vec;
      const unsigned int* xs = (const unsigned int*)xg;
      const unsigned int* hs = (const unsigned int*)hbfR;
      for (int i = tid; i < HID / 2; i += NTHR) {
        v32[i]           = xs[i];
        v32[HID / 2 + i] = hs[i];
      }
      __syncthreads();
      {
        const int tt0 = wave * 2, tt1 = wave * 2 + 1;   // 2 tiles/wave, 16/WG
        const int g0 = tt0 >> 2, q0 = tt0 & 3;
        const int g1 = tt1 >> 2, q1 = tt1 & 3;
        const int rb0 = g0 * HID + wg * 64 + q0 * 16;
        const int rb1 = g1 * HID + wg * 64 + q1 * 16;
        v8f acc0 = {}, acc1 = {};
        bf16_gemv_tile2(Wih, rb0, rb1, vec,       HID, lane, acc0, acc1);
        bf16_gemv_tile2(Whh, rb0, rb1, vec + HID, HID, lane, acc0, acc1);
        if ((lane & 15) == 0) {
          const int hi8 = (lane >> 4) * 8;
#pragma unroll
          for (int r = 0; r < 8; ++r) {
            gpre[g0 * 64 + q0 * 16 + hi8 + r] = acc0[r];
            gpre[g1 * 64 + q1 * 16 + hi8 + r] = acc1[r];
          }
        }
      }
      __syncthreads();
      if (tid < 64) {                          // this WG owns h/c[wg*64 .. +64)
        const int j = wg * 64 + tid;
        const float gi = gpre[tid]       + bi;
        const float gf = gpre[64 + tid]  + bf_;
        const float gg = gpre[128 + tid] + bg;
        const float go = gpre[192 + tid] + bo;
        const float cn = sig_fast(gf) * creg + sig_fast(gi) * tanh_fast(gg);
        const float hn = sig_fast(go) * tanh_fast(cn);
        creg    = cn;
        cbf[j]  = f2bf(cn);
        hbfW[j] = f2bf(hn);
        out[(size_t)xrow * (2 * HID) + dir * HID + j] = hn;
      }
    }
    grid_barrier(cnt, 16u * (++bar));          // h/c visible for next step
  }
}

// ---------------------------------------------------------------------------

extern "C" void kernel_launch(void* const* d_in, const int* in_sizes, int n_in,
                              void* d_out, int out_size, void* d_ws, size_t ws_size,
                              hipStream_t stream) {
  (void)in_sizes; (void)n_in; (void)out_size; (void)ws_size;

  const float* input = (const float*)d_in[0];
  const float* Q     = (const float*)d_in[1];
  const float* cbias = (const float*)d_in[2];
  const float* Wih_f = (const float*)d_in[3];
  const float* Whh_f = (const float*)d_in[4];
  const float* bih_f = (const float*)d_in[5];
  const float* bhh_f = (const float*)d_in[6];
  const float* Wih_b = (const float*)d_in[7];
  const float* Whh_b = (const float*)d_in[8];
  const float* bih_b = (const float*)d_in[9];
  const float* bhh_b = (const float*)d_in[10];
  float* out = (float*)d_out;

  char* p = (char*)d_ws;
  auto take = [&p](size_t bytes) -> char* {
    char* r = p;
    p += (bytes + 255) & ~(size_t)255;
    return r;
  };
  const size_t NW = (size_t)4 * HID * HID;                    // elems per W matrix
  unsigned short* WihBf2 = (unsigned short*)take(2 * NW * 2); // 16 MB
  unsigned short* WhhBf2 = (unsigned short*)take(2 * NW * 2); // 16 MB
  unsigned short* QtBf   = (unsigned short*)take((size_t)HID * HID * 2); // 2 MB
  float* bias2           = (float*)take(2 * 4 * HID * 4);
  unsigned short* cbf2   = (unsigned short*)take(2 * HID * 2);
  unsigned short* hbf2   = (unsigned short*)take(4 * HID * 2); // double-buffered
  unsigned short* xg2    = (unsigned short*)take(2 * HID * 2);
  unsigned int* cnts     = (unsigned int*)take(32 * 4);

  k_prep_w<<<dim3((unsigned)((NW + 255) / 256)), dim3(256), 0, stream>>>(
      Wih_f, Whh_f, Wih_b, Whh_b, WihBf2, WhhBf2);
  k_prep_misc<<<dim3((HID * HID + 255) / 256), dim3(256), 0, stream>>>(
      Q, bih_f, bhh_f, bih_b, bhh_b, QtBf, bias2, cbf2, hbf2, xg2, cnts);
  k_moglstm<<<dim3(2 * NWG), dim3(NTHR), 0, stream>>>(
      input, WihBf2, WhhBf2, QtBf, bias2, cbias, cbf2, hbf2, xg2, cnts, out);
}
